// _Expert_4449586119011
// MI455X (gfx1250) — matrix operand order repaired and checked by reading, not compiled, not run on hardware
//
#include <hip/hip_runtime.h>
#include <stdint.h>

// MoE grouped FFN (E=8, D=1024, H=4096, N=4096, C=512):
//   h = gelu_tanh(x @ w1^T + b1)   per expert   (GEMM1, NT)
//   y = clamp(h @ w2^T + b2, +-0.1) per expert  (GEMM2, NT)
// fp32 in HBM -> v_cvt_pk_bf16_f32 -> v_wmma_f32_16x16x32_bf16 (fp32 accum).
// Out-of-count A rows are loaded from clamped (always-valid) addresses; their
// garbage only reaches accumulator rows the epilogue never stores.

typedef __attribute__((ext_vector_type(4)))  float  v4f;
typedef __attribute__((ext_vector_type(4)))  __bf16 v4bf;
typedef __attribute__((ext_vector_type(8)))  __bf16 v8bf;
typedef __attribute__((ext_vector_type(16))) __bf16 v16bf;
typedef __attribute__((ext_vector_type(8)))  float  v8f;

#define E_ 8
#define D_ 1024
#define H_ 4096
#define N_ 4096
#define C_ 512      // per-expert capacity (N/E)

static __device__ __forceinline__ unsigned short f2bf(float f) {
  return __builtin_bit_cast(unsigned short, (__bf16)f);
}

static __device__ __forceinline__ float gelu_tanh(float x) {
  float x3 = x * x * x;
  return 0.5f * x * (1.0f + tanhf(0.7978845608028654f * (x + 0.044715f * x3)));
}

// ---------------------------------------------------------------------------
// GEMM1: per expert e, rows [0,cnt): h[r, n] = gelu(sum_d x[r,d]*w1[e,n,d] + b1[e,n])
// block tile: 128 rows x 128 cols, K-step 32. 8 waves, wave tile 32x64.
// ---------------------------------------------------------------------------
__global__ __launch_bounds__(256) void moe_gemm1(
    const float* __restrict__ x, const int* __restrict__ counts,
    const float* __restrict__ w1, const float* __restrict__ b1,
    unsigned short* __restrict__ hbuf /* [E][C][H] bf16 */) {

  __shared__ __align__(16) unsigned short ldsA[2][128 * 32];
  __shared__ __align__(16) unsigned short ldsB[2][128 * 32];

  const int e     = blockIdx.z;
  const int nBase = blockIdx.x * 128;   // column tile within H
  const int mTile = blockIdx.y * 128;   // row tile within expert capacity

  int base = 0, cnt = 0;
  for (int i = 0; i < E_; ++i) {
    int c = counts[i];
    if (i < e) base += c;
    if (i == e) cnt = c;
  }
  if (cnt > C_) cnt = C_;

  const int tid  = threadIdx.x;
  const int lane = tid & 31;
  const int wv   = tid >> 5;
  const int m0w  = (wv >> 1) * 32;      // wave row offset (0,32,64,96)
  const int n0w  = (wv & 1) * 64;       // wave col offset (0,64)
  const int r    = lane & 15;
  const int hf   = lane >> 4;

  v8f acc[2][4];
#pragma unroll
  for (int i = 0; i < 2; ++i)
#pragma unroll
    for (int j = 0; j < 4; ++j) acc[i][j] = {};

  const int lrow = tid >> 3;            // 0..31 (4 passes of 32 rows)
  const int lcol = (tid & 7) * 4;       // float4 column group
  const float* wB = w1 + (size_t)e * H_ * D_;

  auto loadTile = [&](int kt, int buf) {
    const int k0 = kt * 32;
    v4f a[4], b[4];
    // phase 1: issue all 8 global loads (no waits in between)
#pragma unroll
    for (int p = 0; p < 4; ++p) {
      const int row = p * 32 + lrow;
      int absRow = base + mTile + row;          // clamp for memory safety only
      if (absRow > N_ - 1) absRow = N_ - 1;
      a[p] = *(const v4f*)(x + (size_t)absRow * D_ + k0 + lcol);
      b[p] = *(const v4f*)(wB + (size_t)(nBase + row) * D_ + k0 + lcol);
    }
    // phase 2: convert + stage to LDS
#pragma unroll
    for (int p = 0; p < 4; ++p) {
      const int row = p * 32 + lrow;
      *(v4bf*)&ldsA[buf][row * 32 + lcol] = __builtin_convertvector(a[p], v4bf);
      *(v4bf*)&ldsB[buf][row * 32 + lcol] = __builtin_convertvector(b[p], v4bf);
    }
  };

  auto compute = [&](int buf) {
    v16bf af[2], bf[4];
#pragma unroll
    for (int i = 0; i < 2; ++i) {
      // A 16x32: lanes 0-15 -> K 0-7 / 16-23, lanes 16-31 -> K 8-15 / 24-31
      const unsigned short* p = &ldsA[buf][(m0w + 16 * i + r) * 32];
      v8bf lo = *(const v8bf*)(p + hf * 8);
      v8bf hi = *(const v8bf*)(p + 16 + hf * 8);
      af[i] = __builtin_shufflevector(lo, hi, 0, 1, 2, 3, 4, 5, 6, 7, 8, 9, 10,
                                      11, 12, 13, 14, 15);
    }
#pragma unroll
    for (int j = 0; j < 4; ++j) {
      // B 32x16: lanes 0-15 -> K 0-7 / 16-23, lanes 16-31 -> K 8-15 / 24-31
      const unsigned short* p = &ldsB[buf][(n0w + 16 * j + r) * 32];
      v8bf lo = *(const v8bf*)(p + hf * 8);
      v8bf hi = *(const v8bf*)(p + 16 + hf * 8);
      bf[j] = __builtin_shufflevector(lo, hi, 0, 1, 2, 3, 4, 5, 6, 7, 8, 9, 10,
                                      11, 12, 13, 14, 15);
    }
#pragma unroll
    for (int i = 0; i < 2; ++i)
#pragma unroll
      for (int j = 0; j < 4; ++j)
        acc[i][j] = __builtin_amdgcn_wmma_f32_16x16x32_bf16(
            false, af[i], false, bf[j], (short)0, acc[i][j], false, false);
  };

  loadTile(0, 0);
  __syncthreads();
  const int KT = D_ / 32;
  for (int kt = 0; kt < KT; ++kt) {
    const int cur = kt & 1;
    if (kt + 1 < KT) loadTile(kt + 1, cur ^ 1);
    compute(cur);
    __syncthreads();
  }

  // epilogue: bias + gelu -> bf16 hbuf (row-count guard lives here)
  const float* bp = b1 + (size_t)e * H_;
#pragma unroll
  for (int j = 0; j < 4; ++j) {
    const int col = nBase + n0w + 16 * j + r;
    const float bias = bp[col];
#pragma unroll
    for (int i = 0; i < 2; ++i) {
#pragma unroll
      for (int v = 0; v < 8; ++v) {
        const int m = m0w + 16 * i + (hf ? 8 : 0) + v;   // C/D layout
        const int grow = mTile + m;
        if (grow < cnt) {
          float val = gelu_tanh(acc[i][j][v] + bias);
          hbuf[((size_t)e * C_ + grow) * H_ + col] = f2bf(val);
        }
      }
    }
  }
}

// ---------------------------------------------------------------------------
// GEMM2: y[r, n] = clamp(sum_h hbuf[e,r,h]*w2[e,n,h] + b2[e,n], +-0.1)
// ---------------------------------------------------------------------------
__global__ __launch_bounds__(256) void moe_gemm2(
    const unsigned short* __restrict__ hbuf, const int* __restrict__ counts,
    const float* __restrict__ w2, const float* __restrict__ b2,
    float* __restrict__ out) {

  __shared__ __align__(16) unsigned short ldsA[2][128 * 32];
  __shared__ __align__(16) unsigned short ldsB[2][128 * 32];

  const int e     = blockIdx.z;
  const int nBase = blockIdx.x * 128;   // column tile within D
  const int mTile = blockIdx.y * 128;

  int base = 0, cnt = 0;
  for (int i = 0; i < E_; ++i) {
    int c = counts[i];
    if (i < e) base += c;
    if (i == e) cnt = c;
  }
  if (cnt > C_) cnt = C_;

  const int tid  = threadIdx.x;
  const int lane = tid & 31;
  const int wv   = tid >> 5;
  const int m0w  = (wv >> 1) * 32;
  const int n0w  = (wv & 1) * 64;
  const int r    = lane & 15;
  const int hf   = lane >> 4;

  v8f acc[2][4];
#pragma unroll
  for (int i = 0; i < 2; ++i)
#pragma unroll
    for (int j = 0; j < 4; ++j) acc[i][j] = {};

  const int lrow = tid >> 3;
  const int lcol = (tid & 7) * 4;
  const unsigned short* hA = hbuf + (size_t)e * C_ * H_;
  const float* wB = w2 + (size_t)e * D_ * H_;

  auto loadTile = [&](int kt, int buf) {
    const int k0 = kt * 32;
    // A: bf16 activations from padded [C][H] slice -> always in-bounds
    const int arow = tid >> 1;
    const int aoff = (tid & 1) * 16;
    const uint4* src =
        (const uint4*)(hA + (size_t)(mTile + arow) * H_ + k0 + aoff);
    uint4 a0 = src[0];
    uint4 a1 = src[1];
    v4f b[4];
#pragma unroll
    for (int p = 0; p < 4; ++p) {
      const int row = p * 32 + lrow;
      b[p] = *(const v4f*)(wB + (size_t)(nBase + row) * H_ + k0 + lcol);
    }
    *(uint4*)&ldsA[buf][arow * 32 + aoff] = a0;
    *(uint4*)&ldsA[buf][arow * 32 + aoff + 8] = a1;
#pragma unroll
    for (int p = 0; p < 4; ++p) {
      const int row = p * 32 + lrow;
      *(v4bf*)&ldsB[buf][row * 32 + lcol] = __builtin_convertvector(b[p], v4bf);
    }
  };

  auto compute = [&](int buf) {
    v16bf af[2], bf[4];
#pragma unroll
    for (int i = 0; i < 2; ++i) {
      const unsigned short* p = &ldsA[buf][(m0w + 16 * i + r) * 32];
      v8bf lo = *(const v8bf*)(p + hf * 8);
      v8bf hi = *(const v8bf*)(p + 16 + hf * 8);
      af[i] = __builtin_shufflevector(lo, hi, 0, 1, 2, 3, 4, 5, 6, 7, 8, 9, 10,
                                      11, 12, 13, 14, 15);
    }
#pragma unroll
    for (int j = 0; j < 4; ++j) {
      const unsigned short* p = &ldsB[buf][(n0w + 16 * j + r) * 32];
      v8bf lo = *(const v8bf*)(p + hf * 8);
      v8bf hi = *(const v8bf*)(p + 16 + hf * 8);
      bf[j] = __builtin_shufflevector(lo, hi, 0, 1, 2, 3, 4, 5, 6, 7, 8, 9, 10,
                                      11, 12, 13, 14, 15);
    }
#pragma unroll
    for (int i = 0; i < 2; ++i)
#pragma unroll
      for (int j = 0; j < 4; ++j)
        acc[i][j] = __builtin_amdgcn_wmma_f32_16x16x32_bf16(
            false, af[i], false, bf[j], (short)0, acc[i][j], false, false);
  };

  loadTile(0, 0);
  __syncthreads();
  const int KT = H_ / 32;
  for (int kt = 0; kt < KT; ++kt) {
    const int cur = kt & 1;
    if (kt + 1 < KT) loadTile(kt + 1, cur ^ 1);
    compute(cur);
    __syncthreads();
  }

  const float* bp = b2 + (size_t)e * D_;
#pragma unroll
  for (int j = 0; j < 4; ++j) {
    const int col = nBase + n0w + 16 * j + r;
    const float bias = bp[col];
#pragma unroll
    for (int i = 0; i < 2; ++i) {
#pragma unroll
      for (int v = 0; v < 8; ++v) {
        const int m = m0w + 16 * i + (hf ? 8 : 0) + v;
        const int grow = mTile + m;
        if (grow < cnt) {
          float val = acc[i][j][v] + bias;
          val = fminf(fmaxf(val, -0.1f), 0.1f);
          out[(size_t)(base + grow) * D_ + col] = val;
        }
      }
    }
  }
}

extern "C" void kernel_launch(void* const* d_in, const int* in_sizes, int n_in,
                              void* d_out, int out_size, void* d_ws,
                              size_t ws_size, hipStream_t stream) {
  (void)in_sizes; (void)n_in; (void)out_size; (void)ws_size;
  const float* inp    = (const float*)d_in[0];
  const int*   counts = (const int*)d_in[1];
  const float* w1     = (const float*)d_in[2];
  const float* b1     = (const float*)d_in[3];
  const float* w2     = (const float*)d_in[4];
  const float* b2     = (const float*)d_in[5];
  float* out = (float*)d_out;
  unsigned short* hbuf = (unsigned short*)d_ws;  // E*C*H bf16 = 32 MiB

  dim3 blk(256);
  dim3 g1(H_ / 128, C_ / 128, E_);   // 32 x 4 x 8 = 1024 blocks
  moe_gemm1<<<g1, blk, 0, stream>>>(inp, counts, w1, b1, hbuf);
  dim3 g2(D_ / 128, C_ / 128, E_);   // 8 x 4 x 8 = 256 blocks
  moe_gemm2<<<g2, blk, 0, stream>>>(hbuf, counts, w2, b2, out);
}
